// OCCNet_438086664223
// MI455X (gfx1250) — matrix operand order repaired and checked by reading, not compiled, not run on hardware
//
#include <hip/hip_runtime.h>
#include <math.h>

typedef __attribute__((ext_vector_type(16))) _Float16 v16h;
typedef __attribute__((ext_vector_type(8)))  float    v8f;

#define NUM_LODS 5
#define IN_DIM   20
#define FSTR     40   // halfs per feature row (80B, multiple of 16B)
#define HSTR     72   // halfs per hidden row (144B, multiple of 16B)
#define WPB      4    // waves per block (128 threads, wave32)

union AFrag { uint4 q[2]; v16h v; };

__device__ __forceinline__ v16h lds_afrag(const _Float16* p0, const _Float16* p1) {
    AFrag a;
    a.q[0] = *(const uint4*)p0;
    a.q[1] = *(const uint4*)p1;
    return a.v;
}

// B-matrix (K x 16 tile) fragment in contiguous K order (same as the A frags):
// lanes 0-15 hold K=kchunk 0..15, lanes 16-31 hold K=16..31 of the 32-row chunk;
// v16h element j corresponds to K = kbase + j, column = col.
__device__ __forceinline__ v16h make_bfrag(const float* __restrict__ W, int col,
                                           int kbase, int kvalid) {
    v16h f;
#pragma unroll
    for (int j = 0; j < 16; ++j) {
        int k = kbase + j;
        float v = (k < kvalid) ? W[k * 64 + col] : 0.0f;
        f[j] = (_Float16)v;
    }
    return f;
}

__device__ __forceinline__ v8f wmma16(v16h a, v16h b, v8f c) {
    return __builtin_amdgcn_wmma_f32_16x16x32_f16(
        /*neg_a=*/false, a, /*neg_b=*/false, b,
        /*c_mod=*/(short)0, c, /*reuse_a=*/false, /*reuse_b=*/false);
}

__global__ __launch_bounds__(128)
void occnet_kernel(const float* __restrict__ x,
                   const float4* __restrict__ g0, const float4* __restrict__ g1,
                   const float4* __restrict__ g2, const float4* __restrict__ g3,
                   const float4* __restrict__ g4,
                   const float* __restrict__ w0, const float* __restrict__ b0,
                   const float* __restrict__ w1, const float* __restrict__ b1,
                   const float* __restrict__ w2, const float* __restrict__ b2,
                   float* __restrict__ out, int npts)
{
    __shared__ _Float16 smF[WPB][32 * FSTR];  // feature tiles (points x 32, K-padded)
    __shared__ _Float16 smH[WPB][32 * HSTR];  // hidden tiles  (points x 64)

    const int lane   = threadIdx.x & 31;
    const int wave   = threadIdx.x >> 5;
    const int half16 = lane & 15;
    const int kbase  = (lane < 16) ? 0 : 16;  // B-frag K base within 32-chunk
    const int abase  = (lane < 16) ? 0 : 16;  // A-frag K base within 32-chunk

    // ---- loop-invariant weight fragments in registers (built once) ----
    v16h B0[4];
#pragma unroll
    for (int n = 0; n < 4; ++n)
        B0[n] = make_bfrag(w0, n * 16 + half16, kbase, IN_DIM);   // rows >=20 zero-padded
    v16h B1[2][4];
#pragma unroll
    for (int kc = 0; kc < 2; ++kc)
#pragma unroll
        for (int n = 0; n < 4; ++n)
            B1[kc][n] = make_bfrag(w1 + kc * 32 * 64, n * 16 + half16, kbase, 32);
    float b0v[4], b1v[4];
#pragma unroll
    for (int n = 0; n < 4; ++n) {
        b0v[n] = b0[n * 16 + half16];
        b1v[n] = b1[n * 16 + half16];
    }
    const float bias2 = b2[0];

    const float4* grids[NUM_LODS] = { g0, g1, g2, g3, g4 };
    const int    res[NUM_LODS]    = { 16, 32, 64, 128, 256 };

    _Float16* fRow = &smF[wave][lane * FSTR];

    const int nGroups = (npts + 31) >> 5;
    for (int grp = blockIdx.x * WPB + wave; grp < nGroups; grp += gridDim.x * WPB) {
        const int p  = grp * 32 + lane;
        const int pc = (p < npts) ? p : (npts - 1);
        const float px = x[pc * 3 + 0];
        const float py = x[pc * 3 + 1];
        const float pz = x[pc * 3 + 2];

        // ---- multi-LOD trilinear feature gather (memory critical path) ----
#pragma unroll
        for (int l = 0; l < NUM_LODS; ++l) {
            const int   r = res[l];
            const float s = 0.5f * (float)(r - 1);
            const float fx = (px + 1.0f) * s;
            const float fy = (py + 1.0f) * s;
            const float fz = (pz + 1.0f) * s;
            int ix = (int)floorf(fx); ix = ix < 0 ? 0 : (ix > r - 2 ? r - 2 : ix);
            int iy = (int)floorf(fy); iy = iy < 0 ? 0 : (iy > r - 2 ? r - 2 : iy);
            int iz = (int)floorf(fz); iz = iz < 0 ? 0 : (iz > r - 2 ? r - 2 : iz);
            const float wx = fx - (float)ix, wy = fy - (float)iy, wz = fz - (float)iz;
            const float wx0 = 1.0f - wx, wy0 = 1.0f - wy, wz0 = 1.0f - wz;

            const int rr  = r * r;
            const int bidx = (ix * r + iy) * r + iz;
            const float4* __restrict__ G = grids[l];

            const float4 c000 = G[bidx];
            const float4 c001 = G[bidx + 1];
            const float4 c010 = G[bidx + r];
            const float4 c011 = G[bidx + r + 1];
            const float4 c100 = G[bidx + rr];
            const float4 c101 = G[bidx + rr + 1];
            const float4 c110 = G[bidx + rr + r];
            const float4 c111 = G[bidx + rr + r + 1];

            const float w000 = wx0 * wy0 * wz0, w001 = wx0 * wy0 * wz;
            const float w010 = wx0 * wy  * wz0, w011 = wx0 * wy  * wz;
            const float w100 = wx  * wy0 * wz0, w101 = wx  * wy0 * wz;
            const float w110 = wx  * wy  * wz0, w111 = wx  * wy  * wz;

            const float f0 = w000*c000.x + w001*c001.x + w010*c010.x + w011*c011.x
                           + w100*c100.x + w101*c101.x + w110*c110.x + w111*c111.x;
            const float f1 = w000*c000.y + w001*c001.y + w010*c010.y + w011*c011.y
                           + w100*c100.y + w101*c101.y + w110*c110.y + w111*c111.y;
            const float f2 = w000*c000.z + w001*c001.z + w010*c010.z + w011*c011.z
                           + w100*c100.z + w101*c101.z + w110*c110.z + w111*c111.z;
            const float f3 = w000*c000.w + w001*c001.w + w010*c010.w + w011*c011.w
                           + w100*c100.w + w101*c101.w + w110*c110.w + w111*c111.w;

            fRow[l * 4 + 0] = (_Float16)f0;
            fRow[l * 4 + 1] = (_Float16)f1;
            fRow[l * 4 + 2] = (_Float16)f2;
            fRow[l * 4 + 3] = (_Float16)f3;
        }
#pragma unroll
        for (int k = IN_DIM; k < 32; ++k) fRow[k] = (_Float16)0.0f;

        // ---- MLP via WMMA: two 16-point tiles per group ----
#pragma unroll
        for (int t = 0; t < 2; ++t) {
            // layer 0: 20(->32) x 64
            const _Float16* ar = &smF[wave][(t * 16 + half16) * FSTR];
            const v16h a0 = lds_afrag(ar + abase, ar + abase + 8);
            v8f h[4];
#pragma unroll
            for (int n = 0; n < 4; ++n) {
                v8f z = { 0.f, 0.f, 0.f, 0.f, 0.f, 0.f, 0.f, 0.f };
                h[n] = wmma16(a0, B0[n], z);
            }
            // bias + ReLU -> LDS hidden (f16, A-layout rows)
            _Float16* hw = &smH[wave][(t * 16) * HSTR];
#pragma unroll
            for (int n = 0; n < 4; ++n)
#pragma unroll
                for (int rr = 0; rr < 8; ++rr) {
                    const int prow = (lane < 16) ? rr : (8 + rr);
                    float v = h[n][rr] + b0v[n];
                    v = v > 0.0f ? v : 0.0f;
                    hw[prow * HSTR + n * 16 + half16] = (_Float16)v;
                }

            // layer 1: 64 x 64 (K split into two chained 32-chunks)
            const _Float16* hr = &smH[wave][(t * 16 + half16) * HSTR];
            const v16h a10 = lds_afrag(hr + abase,      hr + abase + 8);
            const v16h a11 = lds_afrag(hr + 32 + abase, hr + 32 + abase + 8);
            v8f h2[4];
#pragma unroll
            for (int n = 0; n < 4; ++n) {
                v8f z = { 0.f, 0.f, 0.f, 0.f, 0.f, 0.f, 0.f, 0.f };
                v8f acc = wmma16(a10, B1[0][n], z);
                h2[n] = wmma16(a11, B1[1][n], acc);
            }
            // bias + ReLU -> overwrite LDS hidden with layer-1 activations
#pragma unroll
            for (int n = 0; n < 4; ++n)
#pragma unroll
                for (int rr = 0; rr < 8; ++rr) {
                    const int prow = (lane < 16) ? rr : (8 + rr);
                    float v = h2[n][rr] + b1v[n];
                    v = v > 0.0f ? v : 0.0f;
                    hw[prow * HSTR + n * 16 + half16] = (_Float16)v;
                }
        }

        // ---- layer 2 (64 -> 1) as VALU dot + sigmoid, one point per lane ----
        const _Float16* h2r = &smH[wave][lane * HSTR];
        float sacc = bias2;
#pragma unroll
        for (int k = 0; k < 64; ++k)
            sacc += (float)h2r[k] * w2[k];   // w2: uniform address -> scalar cached loads

        if (p < npts)
            out[p] = 1.0f / (1.0f + __expf(-sacc));
    }
}

extern "C" void kernel_launch(void* const* d_in, const int* in_sizes, int n_in,
                              void* d_out, int out_size, void* d_ws, size_t ws_size,
                              hipStream_t stream) {
    (void)n_in; (void)out_size; (void)d_ws; (void)ws_size;
    const float*  x  = (const float*)d_in[0];
    const float4* g0 = (const float4*)d_in[1];
    const float4* g1 = (const float4*)d_in[2];
    const float4* g2 = (const float4*)d_in[3];
    const float4* g3 = (const float4*)d_in[4];
    const float4* g4 = (const float4*)d_in[5];
    const float*  w0 = (const float*)d_in[6];
    const float*  b0 = (const float*)d_in[7];
    const float*  w1 = (const float*)d_in[8];
    const float*  b1 = (const float*)d_in[9];
    const float*  w2 = (const float*)d_in[10];
    const float*  b2 = (const float*)d_in[11];

    const int npts    = in_sizes[0] / 3;
    const int nGroups = (npts + 31) / 32;
    int blocks = (nGroups + WPB - 1) / WPB;
    if (blocks > 2048) blocks = 2048;

    occnet_kernel<<<blocks, 128, 0, stream>>>(x, g0, g1, g2, g3, g4,
                                              w0, b0, w1, b1, w2, b2,
                                              (float*)d_out, npts);
}
